// TransformerBlock_68856915689622
// MI455X (gfx1250) — matrix operand order repaired and checked by reading, not compiled, not run on hardware
//
#include <hip/hip_runtime.h>
#include <hip/hip_bf16.h>
#include <math.h>

// ---------------------------------------------------------------------------
// Types for CDNA5 WMMA (wave32): V_WMMA_F32_16X16X32_F16
// ---------------------------------------------------------------------------
typedef __attribute__((ext_vector_type(16))) _Float16 v16h;
typedef __attribute__((ext_vector_type(8)))  float    v8f;

#define WMMA_F16(a, b, c) \
  __builtin_amdgcn_wmma_f32_16x16x32_f16(false, (a), false, (b), (short)0, (c), false, false)

// ---------------------------------------------------------------------------
// Problem constants
// ---------------------------------------------------------------------------
constexpr int kSeq    = 10;          // tokens per sequence (C)
constexpr int kHd     = 6;
constexpr int kHeads  = 8;
constexpr int kHW     = 4096;        // 64*64
constexpr int kImgPix = 80 * 4096;   // pixels for the conv stage

// fp16 weight-buffer offsets (elements). All K dims padded to multiples of 32.
constexpr size_t OFF_QKVT = 0;        // [2][144][64] (k<48 valid, rest 0)
constexpr size_t OFF_PROJT = 18432;   // [2][48][64]
constexpr size_t OFF_FF1T = 24576;    // [2][192][64]
constexpr size_t OFF_FF2T = 49152;    // [2][48][192]
constexpr size_t OFF_PW1T = 67584;    // [384][96]
constexpr size_t OFF_PW2T = 104448;   // [48][384]
constexpr size_t OFF_SC   = 122880;   // [48][96]
// total 127488 fp16 elements

// ws byte offsets
constexpr size_t WS_Y  = 0;             // f32 y: 80*48*4096*4 = 62914560
constexpr size_t WS_ZN = 62914560ull;   // f16 znorm: 80*4096*96*2 = 62914560
constexpr size_t WS_W  = 125829120ull;  // f16 weights

__device__ __forceinline__ float gelu_exact(float v) {
  return 0.5f * v * (1.0f + erff(v * 0.70710678118654752f));
}

// ---------------------------------------------------------------------------
// WMMA operand builders (ISA 7.12.2 layouts, wave32) — branchless, padded.
// A 16x32 f16: lanes 0-15 rows M=lane (K groups 0..7 & 16..23),
//              lanes 16-31 rows M=lane-16 (K groups 8..15 & 24..31)
// ---------------------------------------------------------------------------
__device__ __forceinline__ v16h load_A_f32(const float* src, int ld, int k0,
                                           int lane) {
  const int row = lane & 15;
  const int kh  = (lane >> 4) * 8;
  const float* p = src + row * ld + k0 + kh;
  v16h a;
#pragma unroll
  for (int i = 0; i < 8; ++i) a[i] = (_Float16)p[i];
#pragma unroll
  for (int i = 0; i < 8; ++i) a[8 + i] = (_Float16)p[16 + i];
  return a;
}

__device__ __forceinline__ v16h load_A_f16(const _Float16* src, int ld, int k0,
                                           int lane) {
  const int row = lane & 15;
  const int kh  = (lane >> 4) * 8;
  const _Float16* p = src + (size_t)row * ld + k0 + kh;
  v16h a;
#pragma unroll
  for (int i = 0; i < 8; ++i) a[i] = p[i];
#pragma unroll
  for (int i = 0; i < 8; ++i) a[8 + i] = p[16 + i];
  return a;
}

// A operand for the skip path, gathered from [channel][4096] f32 planes of x|y.
// k0 is compile-time constant at call sites (fully unrolled) -> branchless.
__device__ __forceinline__ v16h load_A_cat(const float* x, const float* y,
                                           size_t cbase, int hw0, int k0, int lane) {
  const int row = lane & 15;
  const int kh  = (lane >> 4) * 8;
  const int hwp = hw0 + row;
  v16h a;
#pragma unroll
  for (int i = 0; i < 16; ++i) {
    int k = k0 + ((i < 8) ? (kh + i) : (16 + kh + (i - 8)));
    const float* pl = (k < 48) ? (x + cbase + (size_t)k * kHW)
                               : (y + cbase + (size_t)(k - 48) * kHW);
    a[i] = (_Float16)pl[hwp];
  }
  return a;
}

// B 32x16 f16 from pre-transposed, K-padded WT[n][ldk]: K groups kh..kh+7 & 16+kh..16+kh+7 (same order as A).
__device__ __forceinline__ v16h load_B(const _Float16* wt, int ldk, int n0,
                                       int k0, int lane) {
  const int coln = lane & 15;
  const int kh   = (lane >> 4) * 8;
  const _Float16* p = wt + (size_t)(n0 + coln) * ldk + k0 + kh;
  v16h b;
#pragma unroll
  for (int i = 0; i < 8; ++i) b[i] = p[i];
#pragma unroll
  for (int i = 0; i < 8; ++i) b[8 + i] = p[16 + i];
  return b;
}

// ---------------------------------------------------------------------------
// Kernel 1: transpose + convert all weights to fp16 WT[n][kpad] layout
// ---------------------------------------------------------------------------
__global__ void prep_kernel(const float* __restrict__ qkv_w, const float* __restrict__ proj_w,
                            const float* __restrict__ ff1_w, const float* __restrict__ ff2_w,
                            const float* __restrict__ pw1_w, const float* __restrict__ pw2_w,
                            const float* __restrict__ sc_w, _Float16* __restrict__ wbuf) {
  const int tid = blockIdx.x * blockDim.x + threadIdx.x;
  const int nt  = gridDim.x * blockDim.x;
  for (int i = tid; i < 2 * 144 * 64; i += nt) {
    int l = i / 9216, r = i % 9216, n = r / 64, k = r % 64;
    wbuf[OFF_QKVT + i] = (k < 48) ? (_Float16)qkv_w[(l * 48 + k) * 144 + n] : (_Float16)0.f;
  }
  for (int i = tid; i < 2 * 48 * 64; i += nt) {
    int l = i / 3072, r = i % 3072, n = r / 64, k = r % 64;
    wbuf[OFF_PROJT + i] = (k < 48) ? (_Float16)proj_w[(l * 48 + k) * 48 + n] : (_Float16)0.f;
  }
  for (int i = tid; i < 2 * 192 * 64; i += nt) {
    int l = i / 12288, r = i % 12288, n = r / 64, k = r % 64;
    wbuf[OFF_FF1T + i] = (k < 48) ? (_Float16)ff1_w[(l * 48 + k) * 192 + n] : (_Float16)0.f;
  }
  for (int i = tid; i < 2 * 48 * 192; i += nt) {
    int l = i / 9216, r = i % 9216, n = r / 192, k = r % 192;
    wbuf[OFF_FF2T + i] = (_Float16)ff2_w[(l * 192 + k) * 48 + n];
  }
  for (int i = tid; i < 384 * 96; i += nt) {
    int n = i / 96, k = i % 96;
    wbuf[OFF_PW1T + i] = (_Float16)pw1_w[k * 384 + n];
  }
  for (int i = tid; i < 48 * 384; i += nt) {
    int n = i / 384, k = i % 384;
    wbuf[OFF_PW2T + i] = (_Float16)pw2_w[k * 48 + n];
  }
  for (int i = tid; i < 48 * 96; i += nt) wbuf[OFF_SC + i] = (_Float16)sc_w[i];
}

// ---------------------------------------------------------------------------
// Kernel 2: fused 2-layer transformer; 4 waves/block, one wave per sequence,
// 8 sequences per wave. LDS (per wave): t 16x64 f32, qkv/h1 16x192 f16,
// o 16x64 f32. Total 56 KB/block.
// ---------------------------------------------------------------------------
__global__ __launch_bounds__(128) void tf_kernel(
    const float* __restrict__ x, const float* __restrict__ qkv_b,
    const float* __restrict__ proj_b, const float* __restrict__ ff1_b,
    const float* __restrict__ ff2_b, const _Float16* __restrict__ wbuf,
    float* __restrict__ y) {
  __shared__ float    s_t[4][16 * 64];
  __shared__ _Float16 s_qh[4][16 * 192];  // qkv [16][144] f16, then h1 [16][192]
  __shared__ float    s_o[4][16 * 64];

  const int wave = threadIdx.x >> 5;
  const int lane = threadIdx.x & 31;
  float*    tb  = s_t[wave];
  _Float16* qh  = s_qh[wave];
  float*    ob  = s_o[wave];
  const int col = lane & 15;
  const int rbase = (lane >> 4) * 8;

  __builtin_prefetch(wbuf, 0, 1);
  __builtin_prefetch(wbuf + OFF_FF1T, 0, 1);

  for (int it = 0; it < 8; ++it) {
    const int n  = (blockIdx.x * 4 + wave) * 8 + it;
    const int b  = n >> 12;
    const int hw = n & 4095;
    const size_t xbase = (size_t)b * 491520 + hw;  // b*120*4096 + hw

    __syncthreads();
    for (int i = lane; i < 16 * 64; i += 32) {
      int c = i >> 6, f = i & 63;
      tb[i] = (c < kSeq && f < 48) ? x[xbase + (size_t)(c * 48 + f) * kHW] : 0.f;
    }
    __syncthreads();

    for (int l = 0; l < 2; ++l) {
      const _Float16* qkvT  = wbuf + OFF_QKVT  + (size_t)l * 9216;
      const _Float16* projT = wbuf + OFF_PROJT + (size_t)l * 3072;
      const _Float16* ff1T  = wbuf + OFF_FF1T  + (size_t)l * 12288;
      const _Float16* ff2T  = wbuf + OFF_FF2T  + (size_t)l * 9216;

      // --- QKV: (16x48)@(48x144) -> qh f16 [16][144] ---
      {
        v16h a0 = load_A_f32(tb, 64, 0, lane);
        v16h a1 = load_A_f32(tb, 64, 32, lane);
        for (int ntile = 0; ntile < 9; ++ntile) {
          v8f acc = {0, 0, 0, 0, 0, 0, 0, 0};
          acc = WMMA_F16(a0, load_B(qkvT, 64, ntile * 16, 0, lane), acc);
          acc = WMMA_F16(a1, load_B(qkvT, 64, ntile * 16, 32, lane), acc);
#pragma unroll
          for (int r = 0; r < 8; ++r)
            qh[(rbase + r) * 144 + ntile * 16 + col] =
                (_Float16)(acc[r] + qkv_b[l * 144 + ntile * 16 + col]);
        }
      }
      __syncthreads();

      // --- banded attention (VALU): q=cols 48..95, k=0..47, v=96..143 ---
      for (int i = lane; i < 16 * 64; i += 32) ob[i] = 0.f;
      __syncthreads();
      for (int p = lane; p < kHeads * kSeq; p += 32) {
        const int h = p / kSeq, i = p % kSeq;
        const int j0 = (i > 0) ? i - 1 : 0;
        const int j1 = (i < kSeq - 1) ? i + 1 : kSeq - 1;
        float s[3];
        float mx = -3.0e38f;
        for (int j = j0; j <= j1; ++j) {
          float d = 0.f;
          for (int dd = 0; dd < kHd; ++dd)
            d += (float)qh[i * 144 + 48 + h * kHd + dd] *
                 (float)qh[j * 144 + h * kHd + dd];
          s[j - j0] = d * 0.40824829046386302f;  // 1/sqrt(6)
          mx = fmaxf(mx, s[j - j0]);
        }
        float sum = 0.f;
        for (int j = j0; j <= j1; ++j) { s[j - j0] = expf(s[j - j0] - mx); sum += s[j - j0]; }
        const float inv = 1.f / sum;
        for (int dd = 0; dd < kHd; ++dd) {
          float o = 0.f;
          for (int j = j0; j <= j1; ++j)
            o += s[j - j0] * (float)qh[j * 144 + 96 + h * kHd + dd];
          ob[i * 64 + h * kHd + dd] = o * inv;
        }
      }
      __syncthreads();

      // --- proj + residual ---
      {
        v16h a0 = load_A_f32(ob, 64, 0, lane);
        v16h a1 = load_A_f32(ob, 64, 32, lane);
        for (int ntile = 0; ntile < 3; ++ntile) {
          v8f acc = {0, 0, 0, 0, 0, 0, 0, 0};
          acc = WMMA_F16(a0, load_B(projT, 64, ntile * 16, 0, lane), acc);
          acc = WMMA_F16(a1, load_B(projT, 64, ntile * 16, 32, lane), acc);
#pragma unroll
          for (int r = 0; r < 8; ++r)
            tb[(rbase + r) * 64 + ntile * 16 + col] +=
                acc[r] + proj_b[l * 48 + ntile * 16 + col];
        }
      }
      __syncthreads();

      // --- ff1 + GELU -> h1 f16 (reuses qh LDS, [16][192]) ---
      {
        v16h a0 = load_A_f32(tb, 64, 0, lane);
        v16h a1 = load_A_f32(tb, 64, 32, lane);
        for (int ntile = 0; ntile < 12; ++ntile) {
          v8f acc = {0, 0, 0, 0, 0, 0, 0, 0};
          acc = WMMA_F16(a0, load_B(ff1T, 64, ntile * 16, 0, lane), acc);
          acc = WMMA_F16(a1, load_B(ff1T, 64, ntile * 16, 32, lane), acc);
#pragma unroll
          for (int r = 0; r < 8; ++r)
            qh[(rbase + r) * 192 + ntile * 16 + col] =
                (_Float16)gelu_exact(acc[r] + ff1_b[l * 192 + ntile * 16 + col]);
        }
      }
      __syncthreads();

      // --- ff2 + residual: (16x192)@(192x48) ---
      {
        v16h a[6];
#pragma unroll
        for (int kt = 0; kt < 6; ++kt) a[kt] = load_A_f16(qh, 192, kt * 32, lane);
        for (int ntile = 0; ntile < 3; ++ntile) {
          v8f acc = {0, 0, 0, 0, 0, 0, 0, 0};
#pragma unroll
          for (int kt = 0; kt < 6; ++kt)
            acc = WMMA_F16(a[kt], load_B(ff2T, 192, ntile * 16, kt * 32, lane), acc);
#pragma unroll
          for (int r = 0; r < 8; ++r)
            tb[(rbase + r) * 64 + ntile * 16 + col] +=
                acc[r] + ff2_b[l * 48 + ntile * 16 + col];
        }
      }
      __syncthreads();
    }

    // write y in (80,48,4096) layout (same flat index scheme as x)
    for (int i = lane; i < kSeq * 48; i += 32) {
      int c = i / 48, f = i % 48;
      y[xbase + (size_t)(c * 48 + f) * kHW] = tb[c * 64 + f];
    }
  }
}

// ---------------------------------------------------------------------------
// Kernel 3: depthwise 3x3 + bias + LayerNorm(96) -> fp16 znorm[p][96]
// ---------------------------------------------------------------------------
__global__ __launch_bounds__(256) void dwln_kernel(
    const float* __restrict__ x, const float* __restrict__ y,
    const float* __restrict__ dw_w, const float* __restrict__ dw_b,
    const float* __restrict__ ln_g, const float* __restrict__ ln_b,
    _Float16* __restrict__ zn) {
  const int p = blockIdx.x * blockDim.x + threadIdx.x;
  if (p >= kImgPix) return;
  const int bc = p >> 12, hw = p & 4095, h = hw >> 6, w = hw & 63;
  const size_t cbase = (size_t)bc * 48 * kHW;
  float z[96];
  float mu = 0.f;
  for (int c = 0; c < 96; ++c) {
    const float* img = (c < 48) ? (x + cbase + (size_t)c * kHW)
                                : (y + cbase + (size_t)(c - 48) * kHW);
    const float* wv = dw_w + c * 9;
    float acc = 0.f;
#pragma unroll
    for (int dy = -1; dy <= 1; ++dy) {
      int hh = h + dy;
      if (hh < 0 || hh > 63) continue;
#pragma unroll
      for (int dx = -1; dx <= 1; ++dx) {
        int ww = w + dx;
        if (ww < 0 || ww > 63) continue;
        acc += img[hh * 64 + ww] * wv[(dy + 1) * 3 + (dx + 1)];
      }
    }
    z[c] = acc + dw_b[c];
    mu += z[c];
  }
  mu *= (1.f / 96.f);
  float var = 0.f;
  for (int c = 0; c < 96; ++c) { float d = z[c] - mu; var += d * d; }
  const float inv = rsqrtf(var * (1.f / 96.f) + 1e-6f);
  for (int c = 0; c < 96; ++c)
    zn[(size_t)p * 96 + c] = (_Float16)((z[c] - mu) * inv * ln_g[c] + ln_b[c]);
}

// ---------------------------------------------------------------------------
// Kernel 4: pointwise MLP (96->384 GELU ->48) + skip (96->48), WMMA.
// 4 waves/block, one 16-pixel M-tile per wave. LDS 48 KB/block.
// ---------------------------------------------------------------------------
__global__ __launch_bounds__(128) void mlp_kernel(
    const _Float16* __restrict__ zn, const float* __restrict__ x,
    const float* __restrict__ y, const _Float16* __restrict__ wbuf,
    const float* __restrict__ pw1_b, const float* __restrict__ pw2_b,
    const float* __restrict__ sc_b, float* __restrict__ out) {
  __shared__ _Float16 s_h1[4][16 * 384];
  const int wave = threadIdx.x >> 5;
  const int lane = threadIdx.x & 31;
  _Float16* h1 = s_h1[wave];
  const int col = lane & 15;
  const int rbase = (lane >> 4) * 8;
  const int m0 = (blockIdx.x * 4 + wave) * 16;  // 16 pixels, same bc image
  const int bc = m0 >> 12, hw0 = m0 & 4095;
  const size_t cbase = (size_t)bc * 48 * kHW;
  const _Float16* pw1T = wbuf + OFF_PW1T;
  const _Float16* pw2T = wbuf + OFF_PW2T;
  const _Float16* scC  = wbuf + OFF_SC;

  __builtin_prefetch(pw1T, 0, 1);
  __builtin_prefetch(pw2T, 0, 1);

  // pw1: (16x96)@(96x384) + GELU -> h1 f16
  {
    v16h a[3];
#pragma unroll
    for (int kt = 0; kt < 3; ++kt)
      a[kt] = load_A_f16(zn + (size_t)m0 * 96, 96, kt * 32, lane);
    for (int ntile = 0; ntile < 24; ++ntile) {
      v8f acc = {0, 0, 0, 0, 0, 0, 0, 0};
#pragma unroll
      for (int kt = 0; kt < 3; ++kt)
        acc = WMMA_F16(a[kt], load_B(pw1T, 96, ntile * 16, kt * 32, lane), acc);
#pragma unroll
      for (int r = 0; r < 8; ++r)
        h1[(rbase + r) * 384 + ntile * 16 + col] =
            (_Float16)gelu_exact(acc[r] + pw1_b[ntile * 16 + col]);
    }
  }
  __syncthreads();

  // pw2 (16x384)@(384x48) + skip (16x96)@(96x48), fused accumulators
  {
    v16h a2[12];
#pragma unroll
    for (int kt = 0; kt < 12; ++kt) a2[kt] = load_A_f16(h1, 384, kt * 32, lane);
    v16h ac[3];
#pragma unroll
    for (int kt = 0; kt < 3; ++kt) ac[kt] = load_A_cat(x, y, cbase, hw0, kt * 32, lane);

    for (int ntile = 0; ntile < 3; ++ntile) {
      v8f acc = {0, 0, 0, 0, 0, 0, 0, 0};
#pragma unroll
      for (int kt = 0; kt < 12; ++kt)
        acc = WMMA_F16(a2[kt], load_B(pw2T, 384, ntile * 16, kt * 32, lane), acc);
#pragma unroll
      for (int kt = 0; kt < 3; ++kt)
        acc = WMMA_F16(ac[kt], load_B(scC, 96, ntile * 16, kt * 32, lane), acc);
      const int c = ntile * 16 + col;
      const float bias = pw2_b[c] + sc_b[c];
#pragma unroll
      for (int r = 0; r < 8; ++r)
        out[cbase + (size_t)c * kHW + hw0 + rbase + r] = acc[r] + bias;
    }
  }
}

// ---------------------------------------------------------------------------
// Host launcher
// ---------------------------------------------------------------------------
extern "C" void kernel_launch(void* const* d_in, const int* in_sizes, int n_in,
                              void* d_out, int out_size, void* d_ws, size_t ws_size,
                              hipStream_t stream) {
  const float* x      = (const float*)d_in[0];
  const float* qkv_w  = (const float*)d_in[1];
  const float* qkv_b  = (const float*)d_in[2];
  const float* proj_w = (const float*)d_in[3];
  const float* proj_b = (const float*)d_in[4];
  const float* ff1_w  = (const float*)d_in[5];
  const float* ff1_b  = (const float*)d_in[6];
  const float* ff2_w  = (const float*)d_in[7];
  const float* ff2_b  = (const float*)d_in[8];
  const float* dw_w   = (const float*)d_in[9];
  const float* dw_b   = (const float*)d_in[10];
  const float* ln_g   = (const float*)d_in[11];
  const float* ln_b   = (const float*)d_in[12];
  const float* pw1_w  = (const float*)d_in[13];
  const float* pw1_b  = (const float*)d_in[14];
  const float* pw2_w  = (const float*)d_in[15];
  const float* pw2_b  = (const float*)d_in[16];
  const float* sc_w   = (const float*)d_in[17];
  const float* sc_b   = (const float*)d_in[18];

  float* out = (float*)d_out;
  char* ws = (char*)d_ws;
  float* y       = (float*)(ws + WS_Y);
  _Float16* zn   = (_Float16*)(ws + WS_ZN);
  _Float16* wbuf = (_Float16*)(ws + WS_W);

  prep_kernel<<<64, 256, 0, stream>>>(qkv_w, proj_w, ff1_w, ff2_w, pw1_w, pw2_w,
                                      sc_w, wbuf);
  tf_kernel<<<1024, 128, 0, stream>>>(x, qkv_b, proj_b, ff1_b, ff2_b, wbuf, y);
  dwln_kernel<<<kImgPix / 256, 256, 0, stream>>>(x, y, dw_w, dw_b, ln_g, ln_b, zn);
  mlp_kernel<<<kImgPix / 64, 128, 0, stream>>>(zn, x, y, wbuf, pw1_b, pw2_b, sc_b,
                                               out);
}
